// quan_Linear_30889404793387
// MI455X (gfx1250) — compile-verified, matrix operand order checked by reading, not run on hardware
//
#include <hip/hip_runtime.h>
#include <stdint.h>

typedef _Float16 v8h  __attribute__((ext_vector_type(8)));
typedef _Float16 v16h __attribute__((ext_vector_type(16)));
typedef float    v8f  __attribute__((ext_vector_type(8)));
typedef _Float16 h4   __attribute__((ext_vector_type(4)));

#define M_TOT 8192   // B*S = 4*2048
#define N_TOT 4096   // OUT
#define K_TOT 4096   // IN

#define BM 128
#define BN 128
#define BK 32
#define LDT 40       // padded LDS row stride in halves (80B, bank-conflict-free)

// gfx1250 async DMA: global -> LDS, 16B per lane, tracked by ASYNCcnt.
// VDST = per-lane LDS byte offset (low 32 bits of generic LDS pointer),
// VADDR = per-lane 64-bit global address, saddr = off (GV mode).
__device__ __forceinline__ void async_copy_b128(const void* gptr, void* lptr) {
    unsigned       lds  = (unsigned)(uintptr_t)lptr;
    unsigned long long g = (unsigned long long)(uintptr_t)gptr;
    asm volatile("global_load_async_to_lds_b128 %0, %1, off"
                 :: "v"(lds), "v"(g) : "memory");
}
__device__ __forceinline__ void async_wait0() {
    asm volatile("s_wait_asynccnt 0x0" ::: "memory");
}

// ---------------- phase 0: scale = max|w| / 127 ----------------

__global__ void init_amax_kernel(unsigned int* amax) { *amax = 0u; }

__global__ __launch_bounds__(256)
void absmax_kernel(const float* __restrict__ w, unsigned int* __restrict__ amax, int n4) {
    __shared__ float red[256];
    const int tid = threadIdx.x;
    int g = blockIdx.x * 256 + tid;
    const int stride = gridDim.x * 256;
    float m = 0.f;
    for (int i = g; i < n4; i += stride) {
        float4 v = ((const float4*)w)[i];
        m = fmaxf(m, fmaxf(fmaxf(fabsf(v.x), fabsf(v.y)),
                           fmaxf(fabsf(v.z), fabsf(v.w))));
    }
    red[tid] = m;
    __syncthreads();
    for (int s = 128; s > 0; s >>= 1) {
        if (tid < s) red[tid] = fmaxf(red[tid], red[tid + s]);
        __syncthreads();
    }
    if (tid == 0) atomicMax(amax, __float_as_uint(red[0]));  // valid: non-negative floats
}

// ---------------- phase 1: quantize W -> integer-valued f16 (exact) ----------------

__global__ __launch_bounds__(256)
void quant_kernel(const float* __restrict__ w, const unsigned int* __restrict__ amax,
                  _Float16* __restrict__ w16, int n4) {
    const float inv = 127.f / __uint_as_float(*amax);   // 1/scale
    const int i = blockIdx.x * 256 + threadIdx.x;
    if (i < n4) {
        float4 v = ((const float4*)w)[i];
        h4 q;
        q[0] = (_Float16)fminf(fmaxf(rintf(v.x * inv), -127.f), 127.f);
        q[1] = (_Float16)fminf(fmaxf(rintf(v.y * inv), -127.f), 127.f);
        q[2] = (_Float16)fminf(fmaxf(rintf(v.z * inv), -127.f), 127.f);
        q[3] = (_Float16)fminf(fmaxf(rintf(v.w * inv), -127.f), 127.f);
        ((h4*)w16)[i] = q;
    }
}

// ---------------- phase 2: x fp32 -> f16 ----------------

__global__ __launch_bounds__(256)
void cvt_kernel(const float* __restrict__ x, _Float16* __restrict__ x16, int n4) {
    const int i = blockIdx.x * 256 + threadIdx.x;
    if (i < n4) {
        float4 v = ((const float4*)x)[i];
        h4 q;
        q[0] = (_Float16)v.x; q[1] = (_Float16)v.y;
        q[2] = (_Float16)v.z; q[3] = (_Float16)v.w;
        ((h4*)x16)[i] = q;
    }
}

// ---------------- phase 3: WMMA GEMM with async-DMA double buffering ----------------
// A = x16 [M][K] row-major, B = w16 [N][K] row-major (K contiguous: B-fragment friendly)
// 256 threads = 8 wave32, wave grid 4(M) x 2(N); each wave: 32x64 out = 2x4 WMMA tiles.

__global__ __launch_bounds__(256, 1)
void gemm_kernel(const _Float16* __restrict__ A, const _Float16* __restrict__ Bw,
                 const float* __restrict__ bias, const unsigned int* __restrict__ amax,
                 float* __restrict__ out) {
    __shared__ _Float16 As[2][BM][LDT];
    __shared__ _Float16 Bs[2][BN][LDT];

    const int tid   = threadIdx.x;
    const int lane  = tid & 31;
    const int wave  = tid >> 5;
    const int wm    = wave & 3;        // wave row (M)
    const int wn    = wave >> 2;       // wave col (N)
    const int row16 = lane & 15;
    const int hi    = lane >> 4;       // half-wave selector (K sub-block per ISA layout)
    const int mBlock = blockIdx.y * BM;
    const int nBlock = blockIdx.x * BN;

    // this thread's two (row, 16B-segment) slots for tile staging
    const int r0 = tid >> 2,          s0 = (tid & 3) * 8;
    const int r1 = (tid + 256) >> 2,  s1 = ((tid + 256) & 3) * 8;

    v8f zero = {};
    v8f acc[2][4];
#pragma unroll
    for (int i = 0; i < 2; ++i)
#pragma unroll
        for (int j = 0; j < 4; ++j) acc[i][j] = zero;

    // prologue: DMA K-tile 0 into buffer 0
    async_copy_b128(A  + (size_t)(mBlock + r0) * K_TOT + s0, &As[0][r0][s0]);
    async_copy_b128(A  + (size_t)(mBlock + r1) * K_TOT + s1, &As[0][r1][s1]);
    async_copy_b128(Bw + (size_t)(nBlock + r0) * K_TOT + s0, &Bs[0][r0][s0]);
    async_copy_b128(Bw + (size_t)(nBlock + r1) * K_TOT + s1, &Bs[0][r1][s1]);
    async_wait0();
    __syncthreads();

    const int KSTEPS = K_TOT / BK;     // 128
    for (int ks = 0; ks < KSTEPS; ++ks) {
        const int buf = ks & 1;
        const bool pf = (ks + 1 < KSTEPS);
        if (pf) {
            // DMA next K-tile into the other buffer; overlaps with WMMA below
            const size_t kb = (size_t)(ks + 1) * BK;
            async_copy_b128(A  + (size_t)(mBlock + r0) * K_TOT + kb + s0, &As[buf ^ 1][r0][s0]);
            async_copy_b128(A  + (size_t)(mBlock + r1) * K_TOT + kb + s1, &As[buf ^ 1][r1][s1]);
            async_copy_b128(Bw + (size_t)(nBlock + r0) * K_TOT + kb + s0, &Bs[buf ^ 1][r0][s0]);
            async_copy_b128(Bw + (size_t)(nBlock + r1) * K_TOT + kb + s1, &Bs[buf ^ 1][r1][s1]);
        }

        // A/B fragments: lane layout per ISA 16-bit 16x32 matrix
        v16h af[2], bf[4];
#pragma unroll
        for (int mt = 0; mt < 2; ++mt) {
            const _Float16* p = &As[buf][wm * 32 + mt * 16 + row16][hi * 8];
            v8h lo  = *(const v8h*)p;          // K = hi*8 .. hi*8+7
            v8h hi8 = *(const v8h*)(p + 16);   // K = hi*8+16 .. +23
#pragma unroll
            for (int e = 0; e < 8; ++e) { af[mt][e] = lo[e]; af[mt][e + 8] = hi8[e]; }
        }
#pragma unroll
        for (int nt = 0; nt < 4; ++nt) {
            const _Float16* p = &Bs[buf][wn * 64 + nt * 16 + row16][hi * 8];
            v8h lo  = *(const v8h*)p;
            v8h hi8 = *(const v8h*)(p + 16);
#pragma unroll
            for (int e = 0; e < 8; ++e) { bf[nt][e] = lo[e]; bf[nt][e + 8] = hi8[e]; }
        }

#pragma unroll
        for (int mt = 0; mt < 2; ++mt)
#pragma unroll
            for (int nt = 0; nt < 4; ++nt)
                acc[mt][nt] = __builtin_amdgcn_wmma_f32_16x16x32_f16(
                    false, af[mt], false, bf[nt], (short)0, acc[mt][nt], false, false);

        if (pf) async_wait0();
        __syncthreads();
    }

    // epilogue: y = acc * scale + bias   (C/D layout: VGPR r -> M = r + hi*8)
    const float scale = __uint_as_float(*amax) * (1.f / 127.f);
#pragma unroll
    for (int nt = 0; nt < 4; ++nt) {
        const int col = nBlock + wn * 64 + nt * 16 + row16;
        const float bv = bias[col];
#pragma unroll
        for (int mt = 0; mt < 2; ++mt) {
#pragma unroll
            for (int r = 0; r < 8; ++r) {
                const int rowg = mBlock + wm * 32 + mt * 16 + hi * 8 + r;
                out[(size_t)rowg * N_TOT + col] = acc[mt][nt][r] * scale + bv;
            }
        }
    }
}

// ---------------- launch ----------------

extern "C" void kernel_launch(void* const* d_in, const int* in_sizes, int n_in,
                              void* d_out, int out_size, void* d_ws, size_t ws_size,
                              hipStream_t stream) {
    (void)in_sizes; (void)n_in; (void)out_size; (void)ws_size;
    const float* x    = (const float*)d_in[0];
    const float* w    = (const float*)d_in[1];
    const float* bias = (const float*)d_in[2];
    float* out        = (float*)d_out;

    unsigned int* amax = (unsigned int*)d_ws;
    char* base = (char*)d_ws;
    _Float16* w16 = (_Float16*)(base + 256);
    _Float16* x16 = (_Float16*)(base + 256 + (size_t)N_TOT * K_TOT * sizeof(_Float16));

    const int wn4 = (N_TOT * K_TOT) / 4;   // 4,194,304
    const int xn4 = (M_TOT * K_TOT) / 4;   // 8,388,608

    init_amax_kernel<<<1, 1, 0, stream>>>(amax);
    absmax_kernel<<<1024, 256, 0, stream>>>(w, amax, wn4);
    quant_kernel<<<(wn4 + 255) / 256, 256, 0, stream>>>(w, amax, w16, wn4);
    cvt_kernel<<<(xn4 + 255) / 256, 256, 0, stream>>>(x, x16, xn4);

    dim3 grid(N_TOT / BN, M_TOT / BM);     // 32 x 64
    gemm_kernel<<<grid, 256, 0, stream>>>(x16, w16, bias, amax, out);
}
